// ShareNet_64536178590158
// MI455X (gfx1250) — matrix operand order repaired and checked by reading, not compiled, not run on hardware
//
#include <hip/hip_runtime.h>
#include <hip/hip_bf16.h>

#define Bdim 32768
#define Ddim 1024
#define Odim 512

typedef __attribute__((ext_vector_type(16))) __bf16 v16bf;
typedef __attribute__((ext_vector_type(8)))  float  v8f;

union ABFrag {
    v16bf v;
    unsigned u[8];
    uint4 q[2];
};

__device__ __forceinline__ unsigned short f32_to_bf16(float f) {
    unsigned u = __builtin_bit_cast(unsigned, f);
    unsigned r = u + 0x7FFFu + ((u >> 16) & 1u);  // round-to-nearest-even
    return (unsigned short)(r >> 16);
}
__device__ __forceinline__ float bf16_to_f32(unsigned short h) {
    unsigned u = ((unsigned)h) << 16;
    return __builtin_bit_cast(float, u);
}
__device__ __forceinline__ unsigned pack2bf(float a, float b) {
    return (unsigned)f32_to_bf16(a) | ((unsigned)f32_to_bf16(b) << 16);
}

__device__ __forceinline__ v8f wmma_bf16(const ABFrag& a, const ABFrag& b, v8f c) {
    return __builtin_amdgcn_wmma_f32_16x16x32_bf16(false, a.v, false, b.v,
                                                   (short)0, c, false, false);
}

// A-matrix 16x32 bf16, row-major source: lane16 selects M-row:
// lanes 0-15: K0..15 ; lanes 16-31: K16..31
__device__ __forceinline__ void load_frag_rowA(ABFrag& f, const unsigned short* rowPtr,
                                               int k0, int half) {
    const unsigned short* p = rowPtr + k0 + half * 16;
    f.q[0] = *(const uint4*)(p);
    f.q[1] = *(const uint4*)(p + 8);
}
// B-matrix 32x16 bf16, lane16 selects N-column whose data lives in row `rowPtr`
// of a row-major [N,K] matrix: lanes 0-15 take K0..15, lanes 16-31 K16..31
__device__ __forceinline__ void load_frag_rowB(ABFrag& f, const unsigned short* rowPtr,
                                               int k0, int half) {
    const uint4* p = (const uint4*)(rowPtr + k0 + half * 16);
    f.q[0] = p[0];
    f.q[1] = p[1];
}
// A[m][k] = M[b0+k][col]  (column-major gather, stride ldc)
__device__ __forceinline__ void load_fragA_colmajor(ABFrag& f, const unsigned short* base,
                                                    int b0, int col, int half, int ldc) {
#pragma unroll
    for (int t = 0; t < 8; ++t) {
        int k0 = half * 16 + 2 * t;
        unsigned lo = base[(size_t)(b0 + k0) * ldc + col];
        unsigned hi = base[(size_t)(b0 + k0 + 1) * ldc + col];
        f.u[t] = lo | (hi << 16);
    }
}
// B[k][n] = M[b0+k][col] (column-major gather)
__device__ __forceinline__ void load_fragB_colmajor(ABFrag& f, const unsigned short* base,
                                                    int b0, int col, int half, int ldc) {
#pragma unroll
    for (int t = 0; t < 8; ++t) {
        int k0 = half * 16 + 2 * t;
        unsigned lo = base[(size_t)(b0 + k0) * ldc + col];
        unsigned hi = base[(size_t)(b0 + k0 + 1) * ldc + col];
        f.u[t] = lo | (hi << 16);
    }
}

// C/D layout: lanes 0-15 -> M = vgpr, N = lane ; lanes 16-31 -> M = 8+vgpr
__device__ __forceinline__ void store_tile_h(const v8f& c, unsigned short* Hj,
                                             int row0, int col, int half, float bias) {
#pragma unroll
    for (int r = 0; r < 8; ++r)
        Hj[(size_t)(row0 + half * 8 + r) * Odim + col] = f32_to_bf16(c[r] + bias);
}

// ---------------- Stage 1: f32 -> bf16 casts ----------------
__global__ void cvt_f32_bf16_8(const float* __restrict__ src,
                               unsigned short* __restrict__ dst, long n8) {
    long i = (long)blockIdx.x * blockDim.x + threadIdx.x;
    long stride = (long)gridDim.x * blockDim.x;
    for (; i < n8; i += stride) {
        const float4* s = (const float4*)(src + i * 8);
        float4 f0 = s[0], f1 = s[1];
        uint4 o;
        o.x = pack2bf(f0.x, f0.y);
        o.y = pack2bf(f0.z, f0.w);
        o.z = pack2bf(f1.x, f1.y);
        o.w = pack2bf(f1.z, f1.w);
        *(uint4*)(dst + i * 8) = o;
    }
}

// ---------------- Stage 2: H_j = X W_j^T + b (9 GEMMs) ----------------
// Grid: x = O/64 (fastest-varying -> all column groups sharing the same 64 X rows
// are dispatched adjacently, keeping the streamed X slice L2/WGP$-resident),
// y = B/64, z = j. W (9 MB bf16) is fully L2-resident.
__global__ __launch_bounds__(128) void gemm_qkv(const unsigned short* __restrict__ Xbf,
                                                const unsigned short* __restrict__ Wbf,
                                                const float* __restrict__ bias,
                                                unsigned short* __restrict__ Hbf) {
    int j = blockIdx.z;
    int i3 = j / 3;
    int src = (j % 3 == 0) ? i3 : (i3 + 1) % 3;  // {0,1,1, 1,2,2, 2,0,0}
    const unsigned short* X = Xbf + (size_t)src * Bdim * Ddim;
    const unsigned short* W = Wbf + (size_t)j * Odim * Ddim;

    int wave = threadIdx.x >> 5, lane = threadIdx.x & 31;
    int wm = wave >> 1, wn = wave & 1;
    int lane16 = lane & 15, half = lane >> 4;
    int b0 = blockIdx.y * 64 + wm * 32;
    int o0 = blockIdx.x * 64 + wn * 32;

    const unsigned short* pA0 = X + (size_t)(b0 + lane16) * Ddim;
    const unsigned short* pA1 = X + (size_t)(b0 + 16 + lane16) * Ddim;
    const unsigned short* pB0 = W + (size_t)(o0 + lane16) * Ddim;
    const unsigned short* pB1 = W + (size_t)(o0 + 16 + lane16) * Ddim;

    v8f c00 = {}, c01 = {}, c10 = {}, c11 = {};
    for (int d0 = 0; d0 < Ddim; d0 += 32) {
        ABFrag a0, a1, b0f, b1f;
        load_frag_rowA(a0, pA0, d0, half);
        load_frag_rowA(a1, pA1, d0, half);
        load_frag_rowB(b0f, pB0, d0, half);
        load_frag_rowB(b1f, pB1, d0, half);
        c00 = wmma_bf16(a0, b0f, c00);
        c01 = wmma_bf16(a0, b1f, c01);
        c10 = wmma_bf16(a1, b0f, c10);
        c11 = wmma_bf16(a1, b1f, c11);
    }
    float bias0 = bias[(size_t)j * Odim + o0 + lane16];
    float bias1 = bias[(size_t)j * Odim + o0 + 16 + lane16];
    unsigned short* Hj = Hbf + (size_t)j * Bdim * Odim;
    store_tile_h(c00, Hj, b0,      o0 + lane16,      half, bias0);
    store_tile_h(c01, Hj, b0,      o0 + 16 + lane16, half, bias1);
    store_tile_h(c10, Hj, b0 + 16, o0 + lane16,      half, bias0);
    store_tile_h(c11, Hj, b0 + 16, o0 + 16 + lane16, half, bias1);
}

// ---------------- Stage 3: BatchNorm column stats (deterministic) ----------------
// grid (rowGroup=8, colGroup=8, j=9), block 256 = 64 cols x 4 row-lanes
__global__ __launch_bounds__(256) void bn_stats_partial(const unsigned short* __restrict__ Hbf,
                                                        float* __restrict__ partial) {
    int j = blockIdx.z, rg = blockIdx.x, cg = blockIdx.y;
    int c = threadIdx.x & 63, rq = threadIdx.x >> 6;
    int col = cg * 64 + c;
    const unsigned short* Hj = Hbf + (size_t)j * Bdim * Odim;
    float s = 0.f, s2 = 0.f;
    int rend = rg * 4096 + 4096;
    for (int r = rg * 4096 + rq; r < rend; r += 4) {
        float h = bf16_to_f32(Hj[(size_t)r * Odim + col]);
        s += h;
        s2 += h * h;
    }
    __shared__ float ls[256], lq[256];
    ls[threadIdx.x] = s;
    lq[threadIdx.x] = s2;
    __syncthreads();
    if (rq == 0) {
        float S  = ls[c] + ls[c + 64] + ls[c + 128] + ls[c + 192];
        float S2 = lq[c] + lq[c + 64] + lq[c + 128] + lq[c + 192];
        size_t o = (((size_t)j * 8 + rg) * Odim + col) * 2;
        partial[o] = S;
        partial[o + 1] = S2;
    }
}

__global__ void bn_finalize(const float* __restrict__ partial,
                            const float* __restrict__ gammas,
                            const float* __restrict__ betas,
                            float* __restrict__ ss) {
    int idx = blockIdx.x * blockDim.x + threadIdx.x;
    if (idx >= 9 * Odim) return;
    int j = idx / Odim, o = idx % Odim;
    float s = 0.f, s2 = 0.f;
    for (int g = 0; g < 8; ++g) {
        size_t p = (((size_t)j * 8 + g) * Odim + o) * 2;
        s += partial[p];
        s2 += partial[p + 1];
    }
    float mu = s * (1.0f / Bdim);
    float var = s2 * (1.0f / Bdim) - mu * mu;
    float rs = rsqrtf(var + 1e-5f);
    float a = gammas[idx] * rs;
    ss[2 * idx] = a;
    ss[2 * idx + 1] = betas[idx] - mu * a;
}

// ---------------- Stage 4: in-place relu(h*a + c) ----------------
__global__ __launch_bounds__(256) void bn_apply_relu(unsigned short* __restrict__ Hbf,
                                                     const float* __restrict__ ss) {
    const size_t BO = (size_t)Bdim * Odim;        // 2^24
    const size_t words = 9 * BO / 2;
    unsigned* Hw = (unsigned*)Hbf;
    size_t i = (size_t)blockIdx.x * blockDim.x + threadIdx.x;
    size_t stride = (size_t)gridDim.x * blockDim.x;
    for (; i < words; i += stride) {
        size_t e0 = i * 2;
        int j = (int)(e0 >> 24);
        int o = (int)(e0 & (size_t)(Odim - 1));   // even; pair stays in row
        unsigned v = Hw[i];
        float h0 = bf16_to_f32((unsigned short)(v & 0xFFFFu));
        float h1 = bf16_to_f32((unsigned short)(v >> 16));
        size_t si = ((size_t)j * Odim + o) * 2;
        float y0 = fmaxf(h0 * ss[si]     + ss[si + 1], 0.0f);
        float y1 = fmaxf(h1 * ss[si + 2] + ss[si + 3], 0.0f);
        Hw[i] = pack2bf(y0, y1);
    }
}

// ---------------- Stage 5: S_i = Q_i^T K_i / 32 ----------------
// Q,K are 32 MB bf16 each: fully L2-resident, so the per-lane strided
// gathers for the transposed fragments hit L2.
__global__ __launch_bounds__(128) void attn_gemm(const unsigned short* __restrict__ Hbf,
                                                 float* __restrict__ Sbuf) {
    int i = blockIdx.z;
    const unsigned short* Q = Hbf + (size_t)(3 * i)     * Bdim * Odim;
    const unsigned short* K = Hbf + (size_t)(3 * i + 1) * Bdim * Odim;

    int wave = threadIdx.x >> 5, lane = threadIdx.x & 31;
    int wm = wave >> 1, wn = wave & 1;
    int lane16 = lane & 15, half = lane >> 4;
    int m0 = blockIdx.x * 64 + wm * 32;
    int n0 = blockIdx.y * 64 + wn * 32;
    int colA0 = m0 + lane16, colA1 = m0 + 16 + lane16;
    int colB0 = n0 + lane16, colB1 = n0 + 16 + lane16;

    v8f c00 = {}, c01 = {}, c10 = {}, c11 = {};
    for (int b0 = 0; b0 < Bdim; b0 += 32) {
        ABFrag a0, a1, f0, f1;
        load_fragA_colmajor(a0, Q, b0, colA0, half, Odim);
        load_fragA_colmajor(a1, Q, b0, colA1, half, Odim);
        load_fragB_colmajor(f0, K, b0, colB0, half, Odim);
        load_fragB_colmajor(f1, K, b0, colB1, half, Odim);
        c00 = wmma_bf16(a0, f0, c00);
        c01 = wmma_bf16(a0, f1, c01);
        c10 = wmma_bf16(a1, f0, c10);
        c11 = wmma_bf16(a1, f1, c11);
    }
    const float sc = 0.03125f;  // 1/sqrt(1024)
    float* S = Sbuf + (size_t)i * Odim * Odim;
#pragma unroll
    for (int r = 0; r < 8; ++r) {
        int rowA = m0 + half * 8 + r;
        int rowB = m0 + 16 + half * 8 + r;
        S[(size_t)rowA * Odim + n0 + lane16]      = c00[r] * sc;
        S[(size_t)rowA * Odim + n0 + 16 + lane16] = c01[r] * sc;
        S[(size_t)rowB * Odim + n0 + lane16]      = c10[r] * sc;
        S[(size_t)rowB * Odim + n0 + 16 + lane16] = c11[r] * sc;
    }
}

// ---------------- Stage 6: row softmax -> bf16 attn ----------------
__global__ __launch_bounds__(256) void softmax_rows(const float* __restrict__ Sbuf,
                                                    unsigned short* __restrict__ attnB) {
    int row = blockIdx.x;  // [3*512) rows, contiguous
    const float* s = Sbuf + (size_t)row * Odim;
    int t = threadIdx.x;
    float e0 = s[t], e1 = s[t + 256];
    __shared__ float red[256];
    red[t] = fmaxf(e0, e1);
    __syncthreads();
    for (int off = 128; off > 0; off >>= 1) {
        if (t < off) red[t] = fmaxf(red[t], red[t + off]);
        __syncthreads();
    }
    float M = red[0];
    __syncthreads();
    float x0 = __expf(e0 - M), x1 = __expf(e1 - M);
    red[t] = x0 + x1;
    __syncthreads();
    for (int off = 128; off > 0; off >>= 1) {
        if (t < off) red[t] += red[t + off];
        __syncthreads();
    }
    float inv = 1.0f / red[0];
    unsigned short* o = attnB + (size_t)row * Odim;
    o[t] = f32_to_bf16(x0 * inv);
    o[t + 256] = f32_to_bf16(x1 * inv);
}

// ---------------- Stage 7: out_i = V_i attn_i^T ----------------
// Grid: x = O/64 fastest so the 8 column groups sharing the same 64 V rows run
// adjacently (V slice L2-resident); attn (512 KB bf16) is trivially resident.
__global__ __launch_bounds__(128) void out_gemm(const unsigned short* __restrict__ Hbf,
                                                const unsigned short* __restrict__ attnB,
                                                float* __restrict__ out) {
    int i = blockIdx.z;
    const unsigned short* V = Hbf + (size_t)(3 * i + 2) * Bdim * Odim;
    const unsigned short* A = attnB + (size_t)i * Odim * Odim;

    int wave = threadIdx.x >> 5, lane = threadIdx.x & 31;
    int wm = wave >> 1, wn = wave & 1;
    int lane16 = lane & 15, half = lane >> 4;
    int b0 = blockIdx.y * 64 + wm * 32;
    int m0 = blockIdx.x * 64 + wn * 32;

    const unsigned short* pA0 = V + (size_t)(b0 + lane16) * Odim;
    const unsigned short* pA1 = V + (size_t)(b0 + 16 + lane16) * Odim;
    const unsigned short* pB0 = A + (size_t)(m0 + lane16) * Odim;       // attn[m][k] row
    const unsigned short* pB1 = A + (size_t)(m0 + 16 + lane16) * Odim;

    v8f c00 = {}, c01 = {}, c10 = {}, c11 = {};
    for (int kb = 0; kb < Odim; kb += 32) {
        ABFrag a0, a1, b0f, b1f;
        load_frag_rowA(a0, pA0, kb, half);
        load_frag_rowA(a1, pA1, kb, half);
        load_frag_rowB(b0f, pB0, kb, half);
        load_frag_rowB(b1f, pB1, kb, half);
        c00 = wmma_bf16(a0, b0f, c00);
        c01 = wmma_bf16(a0, b1f, c01);
        c10 = wmma_bf16(a1, b0f, c10);
        c11 = wmma_bf16(a1, b1f, c11);
    }
    float* O = out + (size_t)i * Bdim * Odim;
#pragma unroll
    for (int r = 0; r < 8; ++r) {
        int r0 = b0 + half * 8 + r;
        int r1 = b0 + 16 + half * 8 + r;
        O[(size_t)r0 * Odim + m0 + lane16]      = c00[r];
        O[(size_t)r0 * Odim + m0 + 16 + lane16] = c01[r];
        O[(size_t)r1 * Odim + m0 + lane16]      = c10[r];
        O[(size_t)r1 * Odim + m0 + 16 + lane16] = c11[r];
    }
}

extern "C" void kernel_launch(void* const* d_in, const int* in_sizes, int n_in,
                              void* d_out, int out_size, void* d_ws, size_t ws_size,
                              hipStream_t stream) {
    (void)in_sizes; (void)n_in; (void)out_size; (void)ws_size;
    const float* cx = (const float*)d_in[0];
    const float* gx = (const float*)d_in[1];
    const float* wx = (const float*)d_in[2];
    const float* Ws = (const float*)d_in[3];
    const float* bs = (const float*)d_in[4];
    const float* gammas = (const float*)d_in[5];
    const float* betas = (const float*)d_in[6];
    float* out = (float*)d_out;

    char* ws = (char*)d_ws;
    size_t off = 0;
    unsigned short* Xbf = (unsigned short*)(ws + off); off += (size_t)3 * Bdim * Ddim * 2;  // 192 MB
    unsigned short* Wbf = (unsigned short*)(ws + off); off += (size_t)9 * Odim * Ddim * 2;  // 9 MB
    unsigned short* Hbf = (unsigned short*)(ws + off); off += (size_t)9 * Bdim * Odim * 2;  // 288 MB
    float* Sbuf         = (float*)(ws + off);          off += (size_t)3 * Odim * Odim * 4;  // 3 MB
    unsigned short* Atb = (unsigned short*)(ws + off); off += (size_t)3 * Odim * Odim * 2;  // 1.5 MB
    float* partial      = (float*)(ws + off);          off += (size_t)9 * 8 * Odim * 2 * 4; // 288 KB
    float* scsh         = (float*)(ws + off);          off += (size_t)9 * Odim * 2 * 4;     // 36 KB

    // Stage 1: casts
    long n8x = (long)Bdim * Ddim / 8;
    cvt_f32_bf16_8<<<4096, 256, 0, stream>>>(cx, Xbf,                              n8x);
    cvt_f32_bf16_8<<<4096, 256, 0, stream>>>(gx, Xbf + (size_t)1 * Bdim * Ddim,    n8x);
    cvt_f32_bf16_8<<<4096, 256, 0, stream>>>(wx, Xbf + (size_t)2 * Bdim * Ddim,    n8x);
    cvt_f32_bf16_8<<<2304, 256, 0, stream>>>(Ws, Wbf, (long)9 * Odim * Ddim / 8);

    // Stage 2: 9 GEMMs (x = column groups fastest for X L2 reuse)
    gemm_qkv<<<dim3(Odim / 64, Bdim / 64, 9), 128, 0, stream>>>(Xbf, Wbf, bs, Hbf);

    // Stage 3: deterministic BN stats
    bn_stats_partial<<<dim3(8, 8, 9), 256, 0, stream>>>(Hbf, partial);
    bn_finalize<<<18, 256, 0, stream>>>(partial, gammas, betas, scsh);

    // Stage 4: normalize + ReLU in place
    bn_apply_relu<<<4096, 256, 0, stream>>>(Hbf, scsh);

    // Stage 5: S = Q^T K / 32
    attn_gemm<<<dim3(Odim / 64, Odim / 64, 3), 128, 0, stream>>>(Hbf, Sbuf);

    // Stage 6: softmax rows -> bf16
    softmax_rows<<<3 * Odim, 256, 0, stream>>>(Sbuf, Atb);

    // Stage 7: out = V attn^T (directly into d_out, tuple order i=0,1,2)
    out_gemm<<<dim3(Odim / 64, Bdim / 64, 3), 128, 0, stream>>>(Hbf, Atb, out);
}
